// ConvNCF_3891240370410
// MI455X (gfx1250) — matrix operand order repaired and checked by reading, not compiled, not run on hardware
//
#include <hip/hip_runtime.h>
#include <math.h>

// GNN attentive-GRU layer for MI455X (gfx1250), bf16 WMMA path with
// pre-packed bf16 operands (no f32->bf16 conversion in GEMM hot loops).
// V=50000, E=800000, D=H=128 (read generically from in_sizes).

typedef __attribute__((ext_vector_type(16))) __bf16 v16bf;
typedef __attribute__((ext_vector_type(8)))  float  v8f;

__device__ __forceinline__ unsigned short f2bf(float f) {
  unsigned int b = __float_as_uint(f);
  b += 0x7FFFu + ((b >> 16) & 1u);          // round-to-nearest-even
  return (unsigned short)(b >> 16);
}

// One wave computes one 16x16 f32 tile of
//   out = A[row0:row0+16, 0:128] @ W[n0:n0+16, 0:128]^T + bias
// A,W are packed bf16, row-major, leading dim 128 (row stride 256B, so all
// fragment loads below are 16B-aligned). Lane-level operand layouts follow
// CDNA5 ISA §7.12.2 (16-bit A 16x32; B 32x16; 32-bit C/D 16x16).
__device__ __forceinline__ v8f wmma_tile16x16(const unsigned short* __restrict__ A,
                                              const unsigned short* __restrict__ W,
                                              const float* __restrict__ bias,
                                              int row0, int n0, int lane) {
  v8f c;
  const float bn = bias[n0 + (lane & 15)];
#pragma unroll
  for (int r = 0; r < 8; ++r) c[r] = bn;

  const int m  = lane & 15;   // A row within tile / B col within tile
  const int kg = lane >> 4;   // K half selector
  const unsigned short* pa = A + (size_t)(row0 + m) * 128 + kg * 8;
  const unsigned short* pw = W + (size_t)(n0 + m) * 128 + kg * 8;

#pragma unroll
  for (int k0 = 0; k0 < 128; k0 += 32) {
    union { v16bf v; uint4 q[2]; } av, bv;
    av.q[0] = *(const uint4*)(pa + k0);        // K = k0 + kg*8 + (0..7)
    av.q[1] = *(const uint4*)(pa + k0 + 16);   // K = k0 + 16 + kg*8 + (0..7)
    bv.q[0] = *(const uint4*)(pw + k0);        // K = k0 + kg*8 + (0..7)
    bv.q[1] = *(const uint4*)(pw + k0 + 16);   // K = k0 + 16 + kg*8 + (0..7)
    c = __builtin_amdgcn_wmma_f32_16x16x32_bf16(false, av.v, false, bv.v,
                                                (short)0, c, false, false);
  }
  return c;
}

// ---------- f32 -> packed bf16 (4 elems/thread, vectorized) ----------
__global__ void __launch_bounds__(256)
k_f2bf(const float* __restrict__ in, unsigned short* __restrict__ out, int N) {
  const int i = (blockIdx.x * blockDim.x + threadIdx.x) * 4;
  if (i >= N) return;  // N is a multiple of 4 for all arrays here
  const float4 f = *(const float4*)(in + i);
  union { unsigned short s[4]; uint2 u; } o;
  o.s[0] = f2bf(f.x); o.s[1] = f2bf(f.y); o.s[2] = f2bf(f.z); o.s[3] = f2bf(f.w);
  *(uint2*)(out + i) = o.u;
}

// ---------- edge attention: per-node partial dots (wave per node) ----------
__global__ void __launch_bounds__(256)
k_node_partials(const float* __restrict__ feats, const float* __restrict__ Wedge,
                float* __restrict__ adst, float* __restrict__ asrc, int V) {
  const int lane = threadIdx.x & 31;
  const int v = blockIdx.x * 8 + (threadIdx.x >> 5);
  if (v >= V) return;  // wave-uniform exit
  float s1 = 0.f, s2 = 0.f;
  const float* fr = feats + (size_t)v * 128;
#pragma unroll
  for (int t = 0; t < 4; ++t) {
    const int i = lane + t * 32;
    const float x = fr[i];
    s1 += x * Wedge[i];
    s2 += x * Wedge[128 + i];
  }
#pragma unroll
  for (int off = 16; off > 0; off >>= 1) {
    s1 += __shfl_xor(s1, off, 32);
    s2 += __shfl_xor(s2, off, 32);
  }
  if (lane == 0) { adst[v] = s1; asrc[v] = s2; }
}

// ---------- per-edge logit + segment max (ordered-uint atomicMax) ----------
__global__ void __launch_bounds__(256)
k_edge_logit_max(const float* __restrict__ adst, const float* __restrict__ asrc,
                 const float* __restrict__ b_edge, const int* __restrict__ src,
                 const int* __restrict__ dst, float* __restrict__ tbuf,
                 unsigned int* __restrict__ mmax, int E) {
  const int e = blockIdx.x * blockDim.x + threadIdx.x;
  if (e >= E) return;
  const int d = dst[e];
  const float x = adst[d] + asrc[src[e]] + b_edge[0];
  const float t = (x >= 0.f) ? x : 0.01f * x;   // LeakyReLU(0.01)
  tbuf[e] = t;
  const unsigned int bits = __float_as_uint(t);
  const unsigned int u = (bits & 0x80000000u) ? ~bits : (bits | 0x80000000u);
  atomicMax(&mmax[d], u);
}

// ---------- per-edge exp + segment sum ----------
__global__ void __launch_bounds__(256)
k_edge_exp_sum(const int* __restrict__ dst, const unsigned int* __restrict__ mmax,
               float* __restrict__ tbuf, float* __restrict__ ssum, int E) {
  const int e = blockIdx.x * blockDim.x + threadIdx.x;
  if (e >= E) return;
  const int d = dst[e];
  const unsigned int u = mmax[d];
  const unsigned int bits = (u & 0x80000000u) ? (u & 0x7FFFFFFFu) : ~u;
  const float m = __uint_as_float(bits);
  const float w = __expf(tbuf[e] - m);
  tbuf[e] = w;                // reuse buffer: now holds exp weights
  atomicAdd(&ssum[d], w);
}

// ---------- hv = feats @ W_proj^T + b_proj  (WMMA, bf16 operands) ----------
__global__ void __launch_bounds__(256)
k_hv(const unsigned short* __restrict__ featsb, const unsigned short* __restrict__ Wpb,
     const float* __restrict__ bp, float* __restrict__ hv) {
  const int lane = threadIdx.x & 31, wave = threadIdx.x >> 5;
  const int v0 = blockIdx.x * 16;
  const int n0 = wave * 16;                       // 8 waves cover N=128
  v8f c = wmma_tile16x16(featsb, Wpb, bp, v0, n0, lane);
  const int mb = (lane >> 4) * 8, n = n0 + (lane & 15);
#pragma unroll
  for (int r = 0; r < 8; ++r) hv[(size_t)(v0 + mb + r) * 128 + n] = c[r];
}

// ---------- weighted aggregation: c[dst] += a * hv[src]  (wave per edge) ----------
__global__ void __launch_bounds__(256)
k_aggregate(const float* __restrict__ hv, const float* __restrict__ wbuf,
            const float* __restrict__ ssum, const int* __restrict__ src,
            const int* __restrict__ dst, float* __restrict__ c, int E) {
  const int e = blockIdx.x * 8 + (threadIdx.x >> 5);
  if (e >= E) return;  // wave-uniform
  const int lane = threadIdx.x & 31;
  const int s = src[e], d = dst[e];
  const float coeff = wbuf[e] / ssum[d];
  const float* hs = hv + (size_t)s * 128;
  float* cd = c + (size_t)d * 128;
#pragma unroll
  for (int t = 0; t < 4; ++t) {
    const int i = lane + t * 32;
    atomicAdd(&cd[i], hs[i] * coeff);
  }
}

// ---------- context = elu(c), emitted directly as packed bf16 ----------
__global__ void __launch_bounds__(256)
k_elu_bf(const float* __restrict__ c, unsigned short* __restrict__ ctxb, int N) {
  const int i = (blockIdx.x * blockDim.x + threadIdx.x) * 4;
  if (i >= N) return;
  const float4 x = *(const float4*)(c + i);
  union { unsigned short s[4]; uint2 u; } o;
  o.s[0] = f2bf(x.x > 0.f ? x.x : expm1f(x.x));
  o.s[1] = f2bf(x.y > 0.f ? x.y : expm1f(x.y));
  o.s[2] = f2bf(x.z > 0.f ? x.z : expm1f(x.z));
  o.s[3] = f2bf(x.w > 0.f ? x.w : expm1f(x.w));
  *(uint2*)(ctxb + i) = o.u;
}

// ---------- fused GRU: gi/gh via WMMA into LDS, then gates ----------
__global__ void __launch_bounds__(256)
k_gru(const unsigned short* __restrict__ ctxb, const unsigned short* __restrict__ featsb,
      const float* __restrict__ feats,
      const unsigned short* __restrict__ Wihb, const unsigned short* __restrict__ Whhb,
      const float* __restrict__ bih, const float* __restrict__ bhh,
      float* __restrict__ out) {
  __shared__ float sGI[16 * 384];
  __shared__ float sGH[16 * 384];
  const int lane = threadIdx.x & 31, wave = threadIdx.x >> 5;
  const int v0 = blockIdx.x * 16;

  // 24 col-tiles for gi + 24 for gh = 48 jobs over 8 waves (uniform, 6 each)
  for (int job = wave; job < 48; job += 8) {
    const int isGI = (job < 24);
    const int n0 = (isGI ? job : job - 24) * 16;
    v8f c = wmma_tile16x16(isGI ? ctxb : featsb, isGI ? Wihb : Whhb,
                           isGI ? bih : bhh, v0, n0, lane);
    float* s = isGI ? sGI : sGH;
    const int mb = (lane >> 4) * 8, n = n0 + (lane & 15);
#pragma unroll
    for (int r = 0; r < 8; ++r) s[(mb + r) * 384 + n] = c[r];
  }
  __syncthreads();

  for (int t = threadIdx.x; t < 2048; t += 256) {
    const int m = t >> 7, f = t & 127;
    const float* gi = &sGI[m * 384];
    const float* gh = &sGH[m * 384];
    const float rr = 1.f / (1.f + __expf(-(gi[f] + gh[f])));
    const float zz = 1.f / (1.f + __expf(-(gi[128 + f] + gh[128 + f])));
    const float nn = tanhf(gi[256 + f] + rr * gh[256 + f]);
    const float h = feats[(size_t)(v0 + m) * 128 + f];
    const float o = (1.f - zz) * nn + zz * h;
    out[(size_t)(v0 + m) * 128 + f] = fmaxf(o, 0.f);
  }
}

extern "C" void kernel_launch(void* const* d_in, const int* in_sizes, int n_in,
                              void* d_out, int out_size, void* d_ws, size_t ws_size,
                              hipStream_t stream) {
  const float* feats  = (const float*)d_in[0];
  const float* Wedge  = (const float*)d_in[1];
  const float* b_edge = (const float*)d_in[2];
  const float* Wp     = (const float*)d_in[3];
  const float* bp     = (const float*)d_in[4];
  const float* Wih    = (const float*)d_in[5];
  const float* Whh    = (const float*)d_in[6];
  const float* bih    = (const float*)d_in[7];
  const float* bhh    = (const float*)d_in[8];
  const int*   src    = (const int*)d_in[9];
  const int*   dst    = (const int*)d_in[10];

  const int V = in_sizes[0] / 128;   // 50000 (multiple of 16)
  const int E = in_sizes[9];         // 800000
  const int NF = V * 128;
  float* out = (float*)d_out;

  // workspace carve-up (256B aligned chunks)
  char* ws = (char*)d_ws;
  size_t off = 0;
  auto alloc = [&](size_t bytes) { void* p = ws + off; off = (off + bytes + 255) & ~(size_t)255; return p; };
  float*          hv     = (float*)alloc((size_t)NF * sizeof(float));
  float*          cacc   = (float*)alloc((size_t)NF * sizeof(float));        // aggregation accumulator
  unsigned short* featsb = (unsigned short*)alloc((size_t)NF * sizeof(unsigned short));
  unsigned short* ctxb   = (unsigned short*)alloc((size_t)NF * sizeof(unsigned short));
  unsigned short* Wpb    = (unsigned short*)alloc((size_t)128 * 128 * sizeof(unsigned short));
  unsigned short* Wihb   = (unsigned short*)alloc((size_t)384 * 128 * sizeof(unsigned short));
  unsigned short* Whhb   = (unsigned short*)alloc((size_t)384 * 128 * sizeof(unsigned short));
  float*          adst   = (float*)alloc((size_t)V * sizeof(float));
  float*          asrc   = (float*)alloc((size_t)V * sizeof(float));
  unsigned int*   mmax   = (unsigned int*)alloc((size_t)V * sizeof(unsigned int));
  float*          ssum   = (float*)alloc((size_t)V * sizeof(float));
  float*          tw     = (float*)alloc((size_t)E * sizeof(float));
  (void)ws_size;

  // init accumulators (0 in ordered-uint encoding == most-negative float)
  hipMemsetAsync(mmax, 0, (size_t)V * sizeof(unsigned int), stream);
  hipMemsetAsync(ssum, 0, (size_t)V * sizeof(float), stream);
  hipMemsetAsync(cacc, 0, (size_t)NF * sizeof(float), stream);

  // 0) pack bf16 operands once (activations + all GEMM weights)
  k_f2bf<<<(NF / 4 + 255) / 256, 256, 0, stream>>>(feats, featsb, NF);
  k_f2bf<<<(128 * 128 / 4 + 255) / 256, 256, 0, stream>>>(Wp, Wpb, 128 * 128);
  k_f2bf<<<(384 * 128 / 4 + 255) / 256, 256, 0, stream>>>(Wih, Wihb, 384 * 128);
  k_f2bf<<<(384 * 128 / 4 + 255) / 256, 256, 0, stream>>>(Whh, Whhb, 384 * 128);

  // 1) per-node attention partials
  k_node_partials<<<(V + 7) / 8, 256, 0, stream>>>(feats, Wedge, adst, asrc, V);
  // 2) hv projection GEMM (WMMA)
  k_hv<<<V / 16, 256, 0, stream>>>(featsb, Wpb, bp, hv);
  // 3) edge logits + segment max
  k_edge_logit_max<<<(E + 255) / 256, 256, 0, stream>>>(adst, asrc, b_edge, src, dst, tw, mmax, E);
  // 4) exp + segment sum
  k_edge_exp_sum<<<(E + 255) / 256, 256, 0, stream>>>(dst, mmax, tw, ssum, E);
  // 5) weighted aggregation into cacc
  k_aggregate<<<(E + 7) / 8, 256, 0, stream>>>(hv, tw, ssum, src, dst, cacc, E);
  // 6) context = elu(c) -> packed bf16
  k_elu_bf<<<(NF / 4 + 255) / 256, 256, 0, stream>>>(cacc, ctxb, NF);
  // 7) fused GRU cell (two WMMA GEMMs in LDS + gates) -> out
  k_gru<<<V / 16, 256, 0, stream>>>(ctxb, featsb, feats, Wihb, Whhb, bih, bhh, out);
}
